// encoder_chain_91285234909816
// MI455X (gfx1250) — compile-verified, matrix operand order checked by reading, not run on hardware
//
#include <hip/hip_runtime.h>
#include <hip/hip_bf16.h>
#include <stdint.h>

// ---- problem constants (from reference) ----
#define NN  50000      // nodes
#define EE  800000     // edges
#define TT  4          // towers
#define DIN 128
#define DHID 64
#define DOUT 3
#define HH  8          // heads in last GAT

typedef __bf16 bf16_t;
typedef bf16_t v16bf __attribute__((ext_vector_type(16)));
typedef float  v8f   __attribute__((ext_vector_type(8)));

// ---------- small helpers ----------
__device__ __forceinline__ unsigned short f2bf(float f) {
    unsigned int u = __float_as_uint(f);
    u += 0x7fffu + ((u >> 16) & 1u);           // round-to-nearest-even
    return (unsigned short)(u >> 16);
}

__device__ __forceinline__ void atomicMaxF(float* addr, float val) {
    // ordered-int trick; storage initialized to -inf
    if (val >= 0.0f) atomicMax((int*)addr, __float_as_int(val));
    else             atomicMin((unsigned int*)addr, __float_as_uint(val));
}

__device__ __forceinline__ float lrelu(float v) { return v > 0.0f ? v : 0.2f * v; }

// ---------- generic fill ----------
__global__ void fill_k(float* p, float v, size_t n) {
    size_t i = (size_t)blockIdx.x * blockDim.x + threadIdx.x;
    if (i < n) p[i] = v;
}

// ---------- bf16 conversion of x ----------
__global__ void cvt_bf16_k(const float* in, unsigned short* out, size_t n) {
    size_t i = (size_t)blockIdx.x * blockDim.x + threadIdx.x;
    if (i < n) out[i] = f2bf(in[i]);
}

// ---------- pack W1 into WMMA B-operand layout ----------
// wpk layout: [t][kc=4][ct=4][lane=32][16 ushort]
__global__ void packW_k(const float* W1, unsigned short* wpk) {
    int i = blockIdx.x * blockDim.x + threadIdx.x;
    if (i >= TT * 4 * 4 * 32) return;
    int lane = i & 31, ct = (i >> 5) & 3, kc = (i >> 7) & 3, t = i >> 9;
    int col   = ct * 16 + (lane & 15);
    int kbase = kc * 32 + ((lane >> 4) ? 8 : 0);
    unsigned short* dst = wpk + (size_t)i * 16;
    for (int j = 0; j < 8; ++j) {
        dst[j]     = f2bf(W1[((size_t)t * DIN + kbase + j)      * DHID + col]);
        dst[8 + j] = f2bf(W1[((size_t)t * DIN + kbase + 16 + j) * DHID + col]);
    }
}

// ---------- tower mix-weight means ----------
__global__ void towerw_k(const float* fc2, const float* fc3, float* wv) {
    int t = threadIdx.x;
    if (t >= TT) return;
    float s2 = 0.f, s3 = 0.f;
    for (int r = 0; r < TT; ++r) { s2 += fc2[r * TT + t]; s3 += fc3[r * TT + t]; }
    wv[t]     = 1.0f / TT;   // stage-1 mean
    wv[4 + t] = s2 / TT;     // fc2_w.mean(0)
    wv[8 + t] = s3 / TT;     // fc3_w.mean(0)
}

// ---------- stage-1 GEMM via WMMA: feat1[t] = x @ W1[t] ----------
// one wave computes a 16-row x 64-col tile for one tower; K=128 in 4 chunks of 32
__global__ void gemm1_wmma_k(const unsigned short* __restrict__ xbf,
                             const unsigned short* __restrict__ wpk,
                             float* __restrict__ feat1) {
    int gtid = blockIdx.x * blockDim.x + threadIdx.x;
    int wave = gtid >> 5;
    int lane = gtid & 31;
    const int rowTiles = NN / 16;                 // 3125, exact
    if (wave >= TT * rowTiles) return;
    int t  = wave / rowTiles;
    int mt = wave % rowTiles;
    int l15 = lane & 15, hi = lane >> 4;
    int row = mt * 16 + l15;

    v8f acc[4] = {};
    for (int kc = 0; kc < 4; ++kc) {
        union { uint4 q[2]; v16bf v; } A;
        const unsigned short* ap = xbf + (size_t)row * DIN + kc * 32 + hi * 8;
        A.q[0] = *(const uint4*)(ap);
        A.q[1] = *(const uint4*)(ap + 16);
#pragma unroll
        for (int ct = 0; ct < 4; ++ct) {
            union { uint4 q[2]; v16bf v; } B;
            const unsigned short* bp =
                wpk + ((((size_t)t * 4 + kc) * 4 + ct) * 32 + lane) * 16;
            B.q[0] = *(const uint4*)(bp);
            B.q[1] = *(const uint4*)(bp + 8);
            acc[ct] = __builtin_amdgcn_wmma_f32_16x16x32_bf16(
                false, A.v, false, B.v, (short)0, acc[ct], false, false);
        }
    }
#pragma unroll
    for (int ct = 0; ct < 4; ++ct)
#pragma unroll
        for (int r = 0; r < 8; ++r) {
            int orow = mt * 16 + hi * 8 + r;   // C layout: lanes 0-15 rows 0-7, 16-31 rows 8-15
            feat1[((size_t)t * NN + orow) * DHID + ct * 16 + l15] = acc[ct][r];
        }
}

// ---------- el/er per (tower, node, head) ----------
__global__ void elr_k(const float* feat, const float* al, const float* ar,
                      float* el, float* er, int H, int D) {
    size_t total = (size_t)TT * NN * H;
    size_t i = (size_t)blockIdx.x * blockDim.x + threadIdx.x;
    if (i >= total) return;
    int h = (int)(i % H);
    size_t tn = i / H;
    int t = (int)(tn / NN);
    const float* f  = feat + (tn * H + h) * D;
    const float* pa = al + ((size_t)t * H + h) * D;
    const float* pb = ar + ((size_t)t * H + h) * D;
    float s1 = 0.f, s2 = 0.f;
    for (int d = 0; d < D; ++d) { float v = f[d]; s1 += v * pa[d]; s2 += v * pb[d]; }
    el[i] = s1; er[i] = s2;
}

// ---------- edge pass A: segment max of leaky-relu attention logits ----------
__global__ void edgeA_k(const float* el, const float* er, const int* edges,
                        float* m, int H) {
    size_t total = (size_t)TT * EE * H;
    size_t i = (size_t)blockIdx.x * blockDim.x + threadIdx.x;
    if (i >= total) return;
    int h = (int)(i % H);
    size_t te = i / H;
    int e = (int)(te % EE), t = (int)(te / EE);
    int sn = edges[(size_t)t * 2 * EE + e];
    int dn = edges[((size_t)t * 2 + 1) * EE + e];
    float v = lrelu(el[((size_t)t * NN + sn) * H + h] + er[((size_t)t * NN + dn) * H + h]);
    atomicMaxF(&m[((size_t)t * NN + dn) * H + h], v);
}

// m = isfinite(m) ? m : 0
__global__ void fixm_k(float* m, size_t n) {
    size_t i = (size_t)blockIdx.x * blockDim.x + threadIdx.x;
    if (i >= n) return;
    unsigned u = __float_as_uint(m[i]) & 0x7fffffffu;
    if (u >= 0x7f800000u) m[i] = 0.0f;
}

// ---------- edge pass B: segment sum of exp(e - m[dst]) ----------
__global__ void edgeB_k(const float* el, const float* er, const int* edges,
                        const float* m, float* sden, int H) {
    size_t total = (size_t)TT * EE * H;
    size_t i = (size_t)blockIdx.x * blockDim.x + threadIdx.x;
    if (i >= total) return;
    int h = (int)(i % H);
    size_t te = i / H;
    int e = (int)(te % EE), t = (int)(te / EE);
    int sn = edges[(size_t)t * 2 * EE + e];
    int dn = edges[((size_t)t * 2 + 1) * EE + e];
    size_t di = ((size_t)t * NN + dn) * H + h;
    float v = lrelu(el[((size_t)t * NN + sn) * H + h] + er[di]);
    atomicAdd(&sden[di], __expf(v - m[di]));
}

// ---------- edge pass C: out[dst] += wv[t] * softmax * feat[src] ----------
__global__ void edgeC_k(const float* el, const float* er, const float* m,
                        const float* sden, const float* feat, const int* edges,
                        const float* wv, float* out, int H, int D) {
    size_t total = (size_t)TT * EE;
    size_t i = (size_t)blockIdx.x * blockDim.x + threadIdx.x;
    if (i >= total) return;
    int e = (int)(i % EE), t = (int)(i / EE);
    int sn = edges[(size_t)t * 2 * EE + e];
    int dn = edges[((size_t)t * 2 + 1) * EE + e];
    float w = wv[t];
    for (int h = 0; h < H; ++h) {
        size_t si = ((size_t)t * NN + sn) * H + h;
        size_t di = ((size_t)t * NN + dn) * H + h;
        float v = lrelu(el[si] + er[di]);
        float a = __expf(v - m[di]) / sden[di];
        float c = w * a;
        const float* fp = feat + si * D;
        float* op = out + ((size_t)dn * H + h) * D;
        for (int d = 0; d < D; ++d) atomicAdd(&op[d], c * fp[d]);
    }
}

// ---------- out[n,h,d] += sum_t wv[t]*b[t,h,d] ----------
__global__ void bias_k(float* out, const float* b, const float* wv, int H, int D) {
    size_t total = (size_t)NN * H * D;
    size_t i = (size_t)blockIdx.x * blockDim.x + threadIdx.x;
    if (i >= total) return;
    int d = (int)(i % D), h = (int)((i / D) % H);
    float s = 0.f;
    for (int t = 0; t < TT; ++t) s += wv[t] * b[((size_t)t * H + h) * D + d];
    out[i] += s;
}

// ---------- small dense: out[t,n,:Dout] = in[n,:Din] @ W[t] ----------
__global__ void dense_k(const float* in, const float* W, float* out, int Din, int Dout) {
    size_t total = (size_t)TT * NN;
    size_t i = (size_t)blockIdx.x * blockDim.x + threadIdx.x;
    if (i >= total) return;
    int t = (int)(i / NN);
    size_t n = i % NN;
    float acc[24];
    for (int j = 0; j < Dout; ++j) acc[j] = 0.f;
    const float* xr = in + n * Din;
    for (int k = 0; k < Din; ++k) {
        float xv = xr[k];
        const float* wr = W + ((size_t)t * Din + k) * Dout;
        for (int j = 0; j < Dout; ++j) acc[j] += xv * wr[j];
    }
    float* op = out + i * Dout;
    for (int j = 0; j < Dout; ++j) op[j] = acc[j];
}

// ---------- per-(node,head) delta norms ----------
__global__ void nrm_k(const float* hM, float* nrm) {
    size_t total = (size_t)NN * HH;
    size_t i = (size_t)blockIdx.x * blockDim.x + threadIdx.x;
    if (i >= total) return;
    int h = (int)(i % HH);
    size_t n = i / HH;
    float s = 0.f;
    for (int d = 0; d < 3; ++d) {
        float a  = hM[(n * HH + h) * 3 + d];
        float b  = n ? hM[((n - 1) * HH + h) * 3 + d] : 0.0f;
        float dx = a - b;
        s += dx * dx;
    }
    nrm[(size_t)h * NN + n] = sqrtf(s);
}

// ---------- median by bisection, one workgroup per head ----------
__global__ void median_k(const float* nrm, float* dmed) {
    __shared__ float red[256];
    __shared__ int   cnt[256];
    int h = blockIdx.x, tid = threadIdx.x;
    const float* p = nrm + (size_t)h * NN;
    float mx = 0.f;
    for (int n = tid; n < NN; n += 256) mx = fmaxf(mx, p[n]);
    red[tid] = mx; __syncthreads();
    for (int s = 128; s > 0; s >>= 1) {
        if (tid < s) red[tid] = fmaxf(red[tid], red[tid + s]);
        __syncthreads();
    }
    float lo = 0.f, hi = red[0];
    __syncthreads();
    for (int it = 0; it < 32; ++it) {
        float mid = 0.5f * (lo + hi);
        int c = 0;
        for (int n = tid; n < NN; n += 256) c += (p[n] <= mid) ? 1 : 0;
        cnt[tid] = c; __syncthreads();
        for (int s = 128; s > 0; s >>= 1) {
            if (tid < s) cnt[tid] += cnt[tid + s];
            __syncthreads();
        }
        int total = cnt[0]; __syncthreads();
        if (total < NN / 2) lo = mid; else hi = mid;
    }
    if (tid == 0) dmed[h] = 0.5f * (lo + hi) + 1e-4f;
}

// ---------- final: cumsum(dx/med) == hM/med; nan_to_num; +0.3*noise ----------
__global__ void final_k(const float* hM, const float* dmed, const float* noise, float* out) {
    size_t total = (size_t)NN * HH * 3;
    size_t i = (size_t)blockIdx.x * blockDim.x + threadIdx.x;
    if (i >= total) return;
    int h = (int)((i / 3) % HH);
    float v = hM[i] / dmed[h];
    unsigned u = __float_as_uint(v) & 0x7fffffffu;
    if (u > 0x7f800000u)       v = 0.0f;             // NaN
    else if (u == 0x7f800000u) v = (v > 0.f) ? 100.0f : -100.0f;  // +/-inf
    out[i] = v + 0.3f * noise[i];
}

// ===================================================================
extern "C" void kernel_launch(void* const* d_in, const int* in_sizes, int n_in,
                              void* d_out, int out_size, void* d_ws, size_t ws_size,
                              hipStream_t stream) {
    (void)in_sizes; (void)n_in; (void)out_size; (void)ws_size;
    const float* x     = (const float*)d_in[0];
    const int*   edges = (const int*)  d_in[1];
    const float* noise = (const float*)d_in[2];
    const float* W1  = (const float*)d_in[3];
    const float* al1 = (const float*)d_in[4];
    const float* ar1 = (const float*)d_in[5];
    const float* b1  = (const float*)d_in[6];
    const float* W2  = (const float*)d_in[7];
    const float* al2 = (const float*)d_in[8];
    const float* ar2 = (const float*)d_in[9];
    const float* b2  = (const float*)d_in[10];
    const float* W3  = (const float*)d_in[11];
    const float* al3 = (const float*)d_in[12];
    const float* ar3 = (const float*)d_in[13];
    const float* b3  = (const float*)d_in[14];
    const float* WM  = (const float*)d_in[15];
    const float* alM = (const float*)d_in[16];
    const float* arM = (const float*)d_in[17];
    const float* bM  = (const float*)d_in[18];
    const float* fc2 = (const float*)d_in[19];
    const float* fc3 = (const float*)d_in[20];
    float* out = (float*)d_out;

    // ---- bump allocator over d_ws ----
    char* base = (char*)d_ws;
    size_t off = 0;
    auto alloc = [&](size_t bytes) -> char* {
        char* p = base + off;
        off += (bytes + 255) & ~(size_t)255;
        return p;
    };
    unsigned short* xbf  = (unsigned short*)alloc((size_t)NN * DIN * 2);
    unsigned short* wpk  = (unsigned short*)alloc((size_t)TT * 4 * 4 * 32 * 16 * 2);
    float* feat1 = (float*)alloc((size_t)TT * NN * DHID * 4);
    float* elb   = (float*)alloc((size_t)TT * NN * HH * 4);   // sized for max H, reused
    float* erb   = (float*)alloc((size_t)TT * NN * HH * 4);
    float* mb    = (float*)alloc((size_t)TT * NN * HH * 4);
    float* sb    = (float*)alloc((size_t)TT * NN * HH * 4);
    float* h1    = (float*)alloc((size_t)NN * DHID * 4);
    float* feat2 = (float*)alloc((size_t)TT * NN * DOUT * 4);
    float* h2    = (float*)alloc((size_t)NN * DOUT * 4);
    float* feat3 = (float*)alloc((size_t)TT * NN * DOUT * 4);
    float* h3    = (float*)alloc((size_t)NN * DOUT * 4);
    float* featM = (float*)alloc((size_t)TT * NN * HH * DOUT * 4);
    float* hM    = (float*)alloc((size_t)NN * HH * DOUT * 4);
    float* nrmb  = (float*)alloc((size_t)HH * NN * 4);
    float* dmed  = (float*)alloc(HH * 4);
    float* wv    = (float*)alloc(16 * 4);    // wv1 @0, wv2 @4, wv3 @8

    const float NEG_INF = -__builtin_huge_valf();
    auto gr = [](size_t n) { return dim3((unsigned)((n + 255) / 256)); };
    dim3 B(256);

    // ---- prep ----
    cvt_bf16_k<<<gr((size_t)NN * DIN), B, 0, stream>>>(x, xbf, (size_t)NN * DIN);
    packW_k<<<gr(TT * 4 * 4 * 32), B, 0, stream>>>(W1, wpk);
    towerw_k<<<1, 32, 0, stream>>>(fc2, fc3, wv);

    // ---- stage 1: WMMA GEMM + GAT(H=1,D=64), tower-mean into h1 ----
    {
        size_t waves = (size_t)TT * (NN / 16);
        gemm1_wmma_k<<<gr(waves * 32), B, 0, stream>>>(xbf, wpk, feat1);
        elr_k<<<gr((size_t)TT * NN), B, 0, stream>>>(feat1, al1, ar1, elb, erb, 1, DHID);
        fill_k<<<gr((size_t)TT * NN), B, 0, stream>>>(mb, NEG_INF, (size_t)TT * NN);
        fill_k<<<gr((size_t)TT * NN), B, 0, stream>>>(sb, 0.0f, (size_t)TT * NN);
        fill_k<<<gr((size_t)NN * DHID), B, 0, stream>>>(h1, 0.0f, (size_t)NN * DHID);
        edgeA_k<<<gr((size_t)TT * EE), B, 0, stream>>>(elb, erb, edges, mb, 1);
        fixm_k<<<gr((size_t)TT * NN), B, 0, stream>>>(mb, (size_t)TT * NN);
        edgeB_k<<<gr((size_t)TT * EE), B, 0, stream>>>(elb, erb, edges, mb, sb, 1);
        edgeC_k<<<gr((size_t)TT * EE), B, 0, stream>>>(elb, erb, mb, sb, feat1, edges,
                                                       wv, h1, 1, DHID);
        bias_k<<<gr((size_t)NN * DHID), B, 0, stream>>>(h1, b1, wv, 1, DHID);
    }

    // ---- stage 2: dense 64->3, GAT(H=1,D=3), fc2-weighted sum into h2 ----
    {
        dense_k<<<gr((size_t)TT * NN), B, 0, stream>>>(h1, W2, feat2, DHID, DOUT);
        elr_k<<<gr((size_t)TT * NN), B, 0, stream>>>(feat2, al2, ar2, elb, erb, 1, DOUT);
        fill_k<<<gr((size_t)TT * NN), B, 0, stream>>>(mb, NEG_INF, (size_t)TT * NN);
        fill_k<<<gr((size_t)TT * NN), B, 0, stream>>>(sb, 0.0f, (size_t)TT * NN);
        fill_k<<<gr((size_t)NN * DOUT), B, 0, stream>>>(h2, 0.0f, (size_t)NN * DOUT);
        edgeA_k<<<gr((size_t)TT * EE), B, 0, stream>>>(elb, erb, edges, mb, 1);
        fixm_k<<<gr((size_t)TT * NN), B, 0, stream>>>(mb, (size_t)TT * NN);
        edgeB_k<<<gr((size_t)TT * EE), B, 0, stream>>>(elb, erb, edges, mb, sb, 1);
        edgeC_k<<<gr((size_t)TT * EE), B, 0, stream>>>(elb, erb, mb, sb, feat2, edges,
                                                       wv + 4, h2, 1, DOUT);
        bias_k<<<gr((size_t)NN * DOUT), B, 0, stream>>>(h2, b2, wv + 4, 1, DOUT);
    }

    // ---- stage 3: dense 3->3, GAT(H=1,D=3), fc3-weighted sum into h3 ----
    {
        dense_k<<<gr((size_t)TT * NN), B, 0, stream>>>(h2, W3, feat3, DOUT, DOUT);
        elr_k<<<gr((size_t)TT * NN), B, 0, stream>>>(feat3, al3, ar3, elb, erb, 1, DOUT);
        fill_k<<<gr((size_t)TT * NN), B, 0, stream>>>(mb, NEG_INF, (size_t)TT * NN);
        fill_k<<<gr((size_t)TT * NN), B, 0, stream>>>(sb, 0.0f, (size_t)TT * NN);
        fill_k<<<gr((size_t)NN * DOUT), B, 0, stream>>>(h3, 0.0f, (size_t)NN * DOUT);
        edgeA_k<<<gr((size_t)TT * EE), B, 0, stream>>>(elb, erb, edges, mb, 1);
        fixm_k<<<gr((size_t)TT * NN), B, 0, stream>>>(mb, (size_t)TT * NN);
        edgeB_k<<<gr((size_t)TT * EE), B, 0, stream>>>(elb, erb, edges, mb, sb, 1);
        edgeC_k<<<gr((size_t)TT * EE), B, 0, stream>>>(elb, erb, mb, sb, feat3, edges,
                                                       wv + 8, h3, 1, DOUT);
        bias_k<<<gr((size_t)NN * DOUT), B, 0, stream>>>(h3, b3, wv + 8, 1, DOUT);
    }

    // ---- stage M: dense 3->24 (H=8,d=3), GAT(H=8,D=3), fc3-weighted into hM ----
    {
        dense_k<<<gr((size_t)TT * NN), B, 0, stream>>>(h3, WM, featM, DOUT, HH * DOUT);
        elr_k<<<gr((size_t)TT * NN * HH), B, 0, stream>>>(featM, alM, arM, elb, erb, HH, DOUT);
        fill_k<<<gr((size_t)TT * NN * HH), B, 0, stream>>>(mb, NEG_INF, (size_t)TT * NN * HH);
        fill_k<<<gr((size_t)TT * NN * HH), B, 0, stream>>>(sb, 0.0f, (size_t)TT * NN * HH);
        fill_k<<<gr((size_t)NN * HH * DOUT), B, 0, stream>>>(hM, 0.0f, (size_t)NN * HH * DOUT);
        edgeA_k<<<gr((size_t)TT * EE * HH), B, 0, stream>>>(elb, erb, edges, mb, HH);
        fixm_k<<<gr((size_t)TT * NN * HH), B, 0, stream>>>(mb, (size_t)TT * NN * HH);
        edgeB_k<<<gr((size_t)TT * EE * HH), B, 0, stream>>>(elb, erb, edges, mb, sb, HH);
        edgeC_k<<<gr((size_t)TT * EE), B, 0, stream>>>(elb, erb, mb, sb, featM, edges,
                                                       wv + 8, hM, HH, DOUT);
        bias_k<<<gr((size_t)NN * HH * DOUT), B, 0, stream>>>(hM, bM, wv + 8, HH, DOUT);
    }

    // ---- norm_: cumsum(dx/median) == hM/median (telescoping), + noise ----
    nrm_k<<<gr((size_t)NN * HH), B, 0, stream>>>(hM, nrmb);
    median_k<<<HH, 256, 0, stream>>>(nrmb, dmed);
    final_k<<<gr((size_t)NN * HH * DOUT), B, 0, stream>>>(hM, dmed, noise, out);
}
